// Conv2dSubsamplingTsAttRegister_64415919506160
// MI455X (gfx1250) — matrix operand order repaired and checked by reading, not compiled, not run on hardware
//
#include <hip/hip_runtime.h>
#include <hip/hip_bf16.h>

// ---------------------------------------------------------------------------
// Types for CDNA5 WMMA (wave32): D(16x16 f32) = A(16x32 f16) * B(32x16 f16) + C
// ---------------------------------------------------------------------------
typedef __attribute__((ext_vector_type(16))) _Float16 v16h_t;
typedef __attribute__((ext_vector_type(8)))  _Float16 v8h_t;
typedef __attribute__((ext_vector_type(8)))  float    v8f_t;

union AFrag { v16h_t v; v8h_t h[2]; };

// A-operand: 16x32 tile from row-major f16, `ld` = row stride in halves.
// lane L: row M = L&15, half = L>>4; element e -> K = (e/8)*16 + half*8 + (e%8)
__device__ __forceinline__ v16h_t load_a(const _Float16* base, int ld, int lane) {
    int m = lane & 15, half = lane >> 4;
    AFrag f;
    f.h[0] = *(const v8h_t*)(base + (size_t)m * ld + half * 8);
    f.h[1] = *(const v8h_t*)(base + (size_t)m * ld + 16 + half * 8);
    return f.v;
}

// B-operand: 32x16 tile from row-major (KxN) f16, `ld` = row stride in halves.
// lane L: col N = L&15, half = L>>4; element e -> K = (e/8)*16 + half*8 + (e%8)
__device__ __forceinline__ v16h_t load_b(const _Float16* base, int ld, int lane) {
    int n = lane & 15, half = lane >> 4;
    v16h_t r;
    #pragma unroll
    for (int e = 0; e < 16; ++e)
        r[e] = base[(size_t)((e >> 3) * 16 + half * 8 + (e & 7)) * ld + n];
    return r;
}

__device__ __forceinline__ v8f_t wmma_f16(v16h_t a, v16h_t b, v8f_t c) {
    return __builtin_amdgcn_wmma_f32_16x16x32_f16(false, a, false, b, (short)0, c,
                                                  false, false);
}

// Async global -> LDS copy of one 16B chunk per lane (ASYNCcnt-tracked).
__device__ __forceinline__ void async_copy_b128(unsigned lds_addr,
                                                const void* gptr) {
    unsigned long long ga = (unsigned long long)(uintptr_t)gptr;
    asm volatile("global_load_async_to_lds_b128 %0, %1, off"
                 :: "v"(lds_addr), "v"(ga) : "memory");
}

__device__ __forceinline__ void wait_async0() {
    asm volatile("s_wait_asynccnt 0x0" ::: "memory");
}

// ---------------------------------------------------------------------------
// Constants
// ---------------------------------------------------------------------------
#define BB   32
#define TT   2048
#define IDIM 80
#define T1N  1023
#define F1N  39
#define T2N  511
#define T2P  512   // padded
#define F2N  19
#define F2P  32    // padded
#define OD   256
#define SS   256
#define HH   8
#define DK   32

// ---------------------------------------------------------------------------
// Kernel 1: conv1 (8->8, 3x3, stride2) + ReLU.  y1: [B][8][1023][39] f32
// ---------------------------------------------------------------------------
__global__ void conv1_kernel(const float* __restrict__ x,
                             const float* __restrict__ w1,
                             const float* __restrict__ b1,
                             float* __restrict__ y1) {
    size_t idx = (size_t)blockIdx.x * blockDim.x + threadIdx.x;
    const size_t total = (size_t)BB * 8 * T1N * F1N;
    if (idx >= total) return;
    int f1 = idx % F1N; size_t r = idx / F1N;
    int t1 = r % T1N;  r /= T1N;
    int c  = r % 8;    int b = r / 8;
    float acc = b1[c];
    #pragma unroll
    for (int ci = 0; ci < 8; ++ci) {
        const float* xp = x + (((size_t)b * 8 + ci) * TT + 2 * t1) * IDIM + 2 * f1;
        const float* wp = w1 + (c * 8 + ci) * 9;
        #pragma unroll
        for (int kh = 0; kh < 3; ++kh)
            #pragma unroll
            for (int kw = 0; kw < 3; ++kw)
                acc += xp[kh * IDIM + kw] * wp[kh * 3 + kw];
    }
    y1[idx] = fmaxf(acc, 0.f);
}

// ---------------------------------------------------------------------------
// Kernel 2: depthwise conv2 (stride2) + pointwise conv3 + ReLU, write f16
// zero-padded tiles xc: [B*8][512][32]
// ---------------------------------------------------------------------------
__global__ void conv23_kernel(const float* __restrict__ y1,
                              const float* __restrict__ w2,
                              const float* __restrict__ b2,
                              const float* __restrict__ w3,
                              const float* __restrict__ b3,
                              _Float16* __restrict__ xc) {
    int idx = blockIdx.x * blockDim.x + threadIdx.x;  // B * 512 * 32
    if (idx >= BB * T2P * F2P) return;
    int f2 = idx & 31, t2 = (idx >> 5) & (T2P - 1), b = idx >> 14;
    bool valid = (f2 < F2N) && (t2 < T2N);
    float y2v[8];
    #pragma unroll
    for (int c = 0; c < 8; ++c) y2v[c] = 0.f;
    if (valid) {
        #pragma unroll
        for (int c = 0; c < 8; ++c) {
            float a = b2[c];
            const float* yp = y1 + (((size_t)b * 8 + c) * T1N + 2 * t2) * F1N + 2 * f2;
            #pragma unroll
            for (int kh = 0; kh < 3; ++kh)
                #pragma unroll
                for (int kw = 0; kw < 3; ++kw)
                    a += yp[kh * F1N + kw] * w2[c * 9 + kh * 3 + kw];
            y2v[c] = a;
        }
    }
    #pragma unroll
    for (int c = 0; c < 8; ++c) {
        float o = 0.f;
        if (valid) {
            o = b3[c];
            #pragma unroll
            for (int ci = 0; ci < 8; ++ci) o += y2v[ci] * w3[c * 8 + ci];
            o = fmaxf(o, 0.f);
        }
        xc[(((size_t)b * 8 + c) * T2P + t2) * F2P + f2] = (_Float16)o;
    }
}

// ---------------------------------------------------------------------------
// Kernel 3a: f16-transpose wk/wv/wo:  wT[d][n] = w[n][d]
// ---------------------------------------------------------------------------
__global__ void prep_weights_kernel(const float* __restrict__ wk,
                                    const float* __restrict__ wv,
                                    const float* __restrict__ wo,
                                    _Float16* __restrict__ wkT,
                                    _Float16* __restrict__ wvT,
                                    _Float16* __restrict__ woT) {
    int i = blockIdx.x * blockDim.x + threadIdx.x;
    if (i >= OD * OD) return;
    int d = i >> 8, n = i & 255;
    wkT[i] = (_Float16)wk[n * OD + d];
    wvT[i] = (_Float16)wv[n * OD + d];
    woT[i] = (_Float16)wo[n * OD + d];
}

// Kernel 3b: spk_emb f32 -> f16 (row-major kept, used as GEMM A operand)
__global__ void prep_spk_kernel(const float* __restrict__ spk,
                                _Float16* __restrict__ spk16) {
    int i = blockIdx.x * blockDim.x + threadIdx.x;
    if (i < BB * SS * OD) spk16[i] = (_Float16)spk[i];
}

// Kernel 3c: fold out-projection into per-head query weights.
// WfoldT[h][k(32)][j(32)] = sum_n wq[h*32+j][n] * ow[n][k]  (k<19 else 0)
// bfold[h][j] = wq_h . ob + bq[h*32+j]
__global__ void fold_wq_kernel(const float* __restrict__ wq,
                               const float* __restrict__ bq,
                               const float* __restrict__ ow,
                               const float* __restrict__ ob,
                               _Float16* __restrict__ wfoldT,
                               float* __restrict__ bfold) {
    int i = blockIdx.x * blockDim.x + threadIdx.x;  // 8*32*32
    if (i >= HH * 32 * 32) return;
    int h = i >> 10, rem = i & 1023, k = rem >> 5, j = rem & 31;
    float s = 0.f;
    if (k < F2N)
        for (int n = 0; n < OD; ++n) s += wq[(h * 32 + j) * OD + n] * ow[n * F2N + k];
    wfoldT[(h * 32 + k) * 32 + j] = (_Float16)s;
    if (k == 0) {
        float bs = bq[h * 32 + j];
        for (int n = 0; n < OD; ++n) bs += wq[(h * 32 + j) * OD + n] * ob[n];
        bfold[h * 32 + j] = bs;
    }
}

// ---------------------------------------------------------------------------
// Kernel 4: per-head query GEMM (K=32, single WMMA per 16x16 tile)
// q[bh][t][j] = xc[bh][t][:] @ WfoldT[h] + bfold[h][j]
// ---------------------------------------------------------------------------
__global__ void q_kernel(const _Float16* __restrict__ xc,
                         const _Float16* __restrict__ wfoldT,
                         const float* __restrict__ bfold,
                         _Float16* __restrict__ q) {
    int bh = blockIdx.x;                      // 256
    int h = bh & 7;
    int wave = threadIdx.x >> 5, lane = threadIdx.x & 31;
    int m0 = (blockIdx.y * 4 + wave) * 16;    // grid.y=8 -> 32 M tiles
    int n0 = blockIdx.z * 16;                 // grid.z=2
    const _Float16* A  = xc + ((size_t)bh * T2P + m0) * F2P;
    const _Float16* Bp = wfoldT + h * 32 * 32 + n0;
    v16h_t a = load_a(A, F2P, lane);
    v16h_t bf = load_b(Bp, 32, lane);
    v8f_t acc = {};
    acc = wmma_f16(a, bf, acc);
    int col = lane & 15, half = lane >> 4;
    float bias = bfold[h * 32 + n0 + col];
    _Float16* out = q + (size_t)bh * T2P * DK;
    #pragma unroll
    for (int r = 0; r < 8; ++r) {
        int t = m0 + r + half * 8;
        out[(size_t)t * DK + n0 + col] = (_Float16)(acc[r] + bias);
    }
}

// ---------------------------------------------------------------------------
// Kernel 5: k/v projections (K=256 -> 8 WMMAs per tile).
// kT[b][h][j][s]  (transposed: B operand for scores)
// v  [b][h][s][j] (row-major:  B operand for ctx)
// ---------------------------------------------------------------------------
__global__ void kv_kernel(const _Float16* __restrict__ spk16,
                          const _Float16* __restrict__ wkT,
                          const _Float16* __restrict__ wvT,
                          const float* __restrict__ bk,
                          const float* __restrict__ bv,
                          _Float16* __restrict__ kT,
                          _Float16* __restrict__ v_rm) {
    int b = blockIdx.x;                       // 32
    int wave = threadIdx.x >> 5, lane = threadIdx.x & 31;
    int s0 = (blockIdx.y * 4 + wave) * 16;    // grid.y=4 -> 16 S tiles
    int n0 = blockIdx.z * 16;                 // grid.z=16
    const _Float16* A = spk16 + ((size_t)b * SS + s0) * OD;
    int col = lane & 15, half = lane >> 4;
    #pragma unroll
    for (int which = 0; which < 2; ++which) {
        const _Float16* W = which ? wvT : wkT;
        v8f_t acc = {};
        #pragma unroll
        for (int kt = 0; kt < 8; ++kt) {
            v16h_t a  = load_a(A + kt * 32, OD, lane);
            v16h_t bf = load_b(W + (size_t)(kt * 32) * OD + n0, OD, lane);
            acc = wmma_f16(a, bf, acc);
        }
        int n = n0 + col;
        int h = n >> 5, j = n & 31;
        float bias = which ? bv[n] : bk[n];
        #pragma unroll
        for (int r = 0; r < 8; ++r) {
            int s = s0 + r + half * 8;
            _Float16 val = (_Float16)(acc[r] + bias);
            if (which) v_rm[(((size_t)b * HH + h) * SS + s) * DK + j] = val;
            else       kT [(((size_t)b * HH + h) * DK + j) * SS + s] = val;
        }
    }
}

// ---------------------------------------------------------------------------
// Kernel 6: fused scores -> softmax -> ctx, per (b,h), 32 rows per block.
// LDS: 32x256 f32 scores (32 KB, overwritten in place by f16 attn)
//    + 16 KB K^T + 16 KB V staged via async global->LDS (ASYNCcnt).
// ---------------------------------------------------------------------------
__global__ void attn_kernel(const _Float16* __restrict__ q,
                            const _Float16* __restrict__ kT,
                            const _Float16* __restrict__ v_rm,
                            _Float16* __restrict__ ctx) {
    __shared__ float sc[32 * 256];                                   // 32 KB
    __shared__ __attribute__((aligned(32))) _Float16 kTs[32 * 256];  // 16 KB [j][s]
    __shared__ __attribute__((aligned(32))) _Float16 vs[256 * 32];   // 16 KB [s][j]
    int bh = blockIdx.x;                      // 256
    int rowbase = blockIdx.y * 32;            // grid.y=16
    int wave = threadIdx.x >> 5, lane = threadIdx.x & 31;
    int col = lane & 15, half = lane >> 4;
    const float scale = 0.17677669529663687f; // 1/sqrt(32)

    // ---- stage K^T and V into LDS asynchronously (once per block) ----
    {
        const _Float16* gk = kT   + (size_t)bh * DK * SS;
        const _Float16* gv = v_rm + (size_t)bh * SS * DK;
        unsigned lk = (unsigned)(uintptr_t)&kTs[0];
        unsigned lv = (unsigned)(uintptr_t)&vs[0];
        #pragma unroll
        for (int i = 0; i < 8; ++i) {
            unsigned off = ((unsigned)threadIdx.x + i * 128u) * 8u;  // halves
            async_copy_b128(lk + off * 2u, gk + off);
            async_copy_b128(lv + off * 2u, gv + off);
        }
    }

    // overlap with staging: load both Q row-tile A fragments from global
    const _Float16* Q = q + ((size_t)bh * T2P + rowbase) * DK;
    v16h_t aq0 = load_a(Q, DK, lane);
    v16h_t aq1 = load_a(Q + 16 * DK, DK, lane);

    wait_async0();
    __syncthreads();

    // ---- scores = Q K^T / sqrt(dk): wave w covers s-tiles [4w, 4w+4) ----
    #pragma unroll
    for (int i = 0; i < 4; ++i) {
        int st = wave * 4 + i;
        v16h_t bf = load_b(&kTs[st * 16], SS, lane);
        v8f_t acc0 = {}, acc1 = {};
        acc0 = wmma_f16(aq0, bf, acc0);
        acc1 = wmma_f16(aq1, bf, acc1);
        #pragma unroll
        for (int r = 0; r < 8; ++r) {
            sc[(r + half * 8) * 256 + st * 16 + col]      = acc0[r] * scale;
            sc[(16 + r + half * 8) * 256 + st * 16 + col] = acc1[r] * scale;
        }
    }
    __syncthreads();

    // ---- row softmax (one thread per row), write f16 attn in place ----
    if (threadIdx.x < 32) {
        float* row = &sc[threadIdx.x * 256];
        float mx = -1e30f;
        for (int s = 0; s < 256; ++s) mx = fmaxf(mx, row[s]);
        float sum = 0.f;
        for (int s = 0; s < 256; ++s) { float e = __expf(row[s] - mx); row[s] = e; sum += e; }
        float inv = 1.0f / sum;
        _Float16* rh = (_Float16*)row;        // in-place: 2s < 4s, safe serially
        for (int s = 0; s < 256; ++s) rh[s] = (_Float16)(row[s] * inv);
    }
    __syncthreads();

    // ---- ctx = attn @ V: wave w owns (mt = w>>1, nt = w&1) ----
    const _Float16* attn_h = (const _Float16*)sc;   // row stride 512 halves
    int mt = wave >> 1, nt = wave & 1;
    v8f_t acc = {};
    #pragma unroll
    for (int kt = 0; kt < 8; ++kt) {
        v16h_t a2 = load_a(attn_h + (size_t)(mt * 16) * 512 + kt * 32, 512, lane);
        v16h_t b2 = load_b(&vs[(size_t)(kt * 32) * DK + nt * 16], DK, lane);
        acc = wmma_f16(a2, b2, acc);
    }
    int hh = bh & 7, b = bh >> 3;
    #pragma unroll
    for (int r = 0; r < 8; ++r) {
        int t = rowbase + mt * 16 + r + half * 8;
        ctx[((size_t)b * T2P + t) * OD + hh * DK + nt * 16 + col] = (_Float16)acc[r];
    }
}

// ---------------------------------------------------------------------------
// Kernel 7: y = (ctx @ wo^T + bo) * 16 + posenc(t, n), store rows t < 511
// ---------------------------------------------------------------------------
__global__ void out_kernel(const _Float16* __restrict__ ctx,
                           const _Float16* __restrict__ woT,
                           const float* __restrict__ bo,
                           float* __restrict__ yout) {
    int b = blockIdx.x;                       // 32
    int wave = threadIdx.x >> 5, lane = threadIdx.x & 31;
    int m0 = (blockIdx.y * 4 + wave) * 16;    // grid.y=8
    int n0 = blockIdx.z * 16;                 // grid.z=16
    const _Float16* A = ctx + ((size_t)b * T2P + m0) * OD;
    v8f_t acc = {};
    #pragma unroll
    for (int kt = 0; kt < 8; ++kt) {
        v16h_t a  = load_a(A + kt * 32, OD, lane);
        v16h_t bf = load_b(woT + (size_t)(kt * 32) * OD + n0, OD, lane);
        acc = wmma_f16(a, bf, acc);
    }
    int col = lane & 15, half = lane >> 4;
    int n = n0 + col;
    float bias16 = bo[n] * 16.0f;
    float freq = __expf(-(float)(n & ~1) * (9.210340371976184f / 256.0f));
    #pragma unroll
    for (int r = 0; r < 8; ++r) {
        int t = m0 + r + half * 8;
        if (t < T2N) {
            float pe = (n & 1) ? __cosf((float)t * freq) : __sinf((float)t * freq);
            yout[((size_t)b * T2N + t) * OD + n] = acc[r] * 16.0f + bias16 + pe;
        }
    }
}

// Kernel 8: mask tail (all-true -> 1.0)
__global__ void mask_kernel(float* __restrict__ out, int count) {
    int i = blockIdx.x * blockDim.x + threadIdx.x;
    if (i < count) out[i] = 1.0f;
}

// ---------------------------------------------------------------------------
// Launch
// ---------------------------------------------------------------------------
extern "C" void kernel_launch(void* const* d_in, const int* in_sizes, int n_in,
                              void* d_out, int out_size, void* d_ws, size_t ws_size,
                              hipStream_t stream) {
    const float* x   = (const float*)d_in[0];
    const float* spk = (const float*)d_in[2];
    const float* w1 = (const float*)d_in[4];  const float* b1 = (const float*)d_in[5];
    const float* w2 = (const float*)d_in[6];  const float* b2 = (const float*)d_in[7];
    const float* w3 = (const float*)d_in[8];  const float* b3 = (const float*)d_in[9];
    const float* ow = (const float*)d_in[10]; const float* ob = (const float*)d_in[11];
    const float* wq = (const float*)d_in[12]; const float* bq = (const float*)d_in[13];
    const float* wk = (const float*)d_in[14]; const float* bk = (const float*)d_in[15];
    const float* wv = (const float*)d_in[16]; const float* bv = (const float*)d_in[17];
    const float* wo = (const float*)d_in[18]; const float* bo = (const float*)d_in[19];

    char* ws = (char*)d_ws;
    auto alloc = [&](size_t bytes) -> char* {
        char* p = ws; ws += (bytes + 255) & ~(size_t)255; return p;
    };
    float*    y1     = (float*)   alloc((size_t)BB * 8 * T1N * F1N * 4);
    _Float16* xc     = (_Float16*)alloc((size_t)BB * 8 * T2P * F2P * 2);
    _Float16* wkT    = (_Float16*)alloc((size_t)OD * OD * 2);
    _Float16* wvT    = (_Float16*)alloc((size_t)OD * OD * 2);
    _Float16* woT    = (_Float16*)alloc((size_t)OD * OD * 2);
    _Float16* wfoldT = (_Float16*)alloc((size_t)HH * 32 * 32 * 2);
    float*    bfold  = (float*)   alloc((size_t)HH * 32 * 4);
    _Float16* spk16  = (_Float16*)alloc((size_t)BB * SS * OD * 2);
    _Float16* qbuf   = (_Float16*)alloc((size_t)BB * HH * T2P * DK * 2);
    _Float16* kT     = (_Float16*)alloc((size_t)BB * HH * DK * SS * 2);
    _Float16* v_rm   = (_Float16*)alloc((size_t)BB * HH * SS * DK * 2);
    _Float16* ctx    = (_Float16*)alloc((size_t)BB * T2P * OD * 2);

    // prep (independent of conv path)
    prep_weights_kernel<<<dim3((OD * OD + 255) / 256), dim3(256), 0, stream>>>(
        wk, wv, wo, wkT, wvT, woT);
    prep_spk_kernel<<<dim3((BB * SS * OD + 255) / 256), dim3(256), 0, stream>>>(
        spk, spk16);
    fold_wq_kernel<<<dim3((HH * 32 * 32 + 255) / 256), dim3(256), 0, stream>>>(
        wq, bq, ow, ob, wfoldT, bfold);

    // conv subsampling
    {
        size_t total = (size_t)BB * 8 * T1N * F1N;
        conv1_kernel<<<dim3((unsigned)((total + 255) / 256)), dim3(256), 0, stream>>>(
            x, w1, b1, y1);
    }
    conv23_kernel<<<dim3((BB * T2P * F2P + 255) / 256), dim3(256), 0, stream>>>(
        y1, w2, b2, w3, b3, xc);

    // projections
    q_kernel <<<dim3(BB * HH, 8, 2),  dim3(128), 0, stream>>>(xc, wfoldT, bfold, qbuf);
    kv_kernel<<<dim3(BB, 4, 16),      dim3(128), 0, stream>>>(spk16, wkT, wvT, bk, bv,
                                                              kT, v_rm);
    // attention (async-staged K^T/V in LDS)
    attn_kernel<<<dim3(BB * HH, 16), dim3(128), 0, stream>>>(qbuf, kT, v_rm, ctx);

    // output projection + posenc
    out_kernel<<<dim3(BB, 8, 16), dim3(128), 0, stream>>>(ctx, woT, bo, (float*)d_out);

    // mask output (all true)
    const int yElems = BB * T2N * OD;  // 4,186,112
    if (out_size > yElems) {
        int cnt = out_size - yElems;
        mask_kernel<<<dim3((cnt + 255) / 256), dim3(256), 0, stream>>>(
            (float*)d_out + yElems, cnt);
    }
}
